// CausalGraphLearning_89077621719712
// MI455X (gfx1250) — matrix operand order repaired and checked by reading, not compiled, not run on hardware
//
#include <hip/hip_runtime.h>
#include <math.h>

typedef __attribute__((ext_vector_type(16))) _Float16 v16h;
typedef __attribute__((ext_vector_type(8)))  _Float16 v8h;
typedef __attribute__((ext_vector_type(4)))  _Float16 v4h;
typedef __attribute__((ext_vector_type(8)))  float    v8f;

#define BATCH 8
#define NNODE 256
#define FDIM  512
#define HDIM  256
#define PS_PAD 260   // 256 + 4 floats: avoids 64-bank LDS conflicts on stride-256 rows

#if defined(__has_builtin) && __has_builtin(__builtin_amdgcn_sched_group_barrier)
#define SCHED_GROUP(mask, size, id) __builtin_amdgcn_sched_group_barrier((mask), (size), (id))
#else
#define SCHED_GROUP(mask, size, id)
#endif

__device__ __forceinline__ float sigmoidf_(float x) { return 1.0f / (1.0f + __expf(-x)); }

// A fragment: 16 contiguous halves (K contiguous, matches B fragment)
__device__ __forceinline__ v16h load_afrag(const _Float16* __restrict__ p)
{
    const v8h lo = *(const v8h*)p;
    const v8h hi = *(const v8h*)(p + 8);
    return __builtin_shufflevector(lo, hi,
        0, 1, 2, 3, 4, 5, 6, 7, 8, 9, 10, 11, 12, 13, 14, 15);
}

// B fragment: 16 contiguous halves (pre-transposed weights, K contiguous)
__device__ __forceinline__ v16h load_bfrag(const _Float16* __restrict__ p)
{
    const v8h lo = *(const v8h*)p;
    const v8h hi = *(const v8h*)(p + 8);
    return __builtin_shufflevector(lo, hi,
        0, 1, 2, 3, 4, 5, 6, 7, 8, 9, 10, 11, 12, 13, 14, 15);
}

// ---------------------------------------------------------------------------
// f32 -> f16 convert (vectorized float4 -> half4), n % 4 == 0
// ---------------------------------------------------------------------------
__global__ __launch_bounds__(256)
void cvt_f16_kernel(const float* __restrict__ src, _Float16* __restrict__ dst, long n)
{
    const long i = ((long)blockIdx.x * 256 + threadIdx.x) * 4;
    if (i < n) {
        const float4 v = *(const float4*)(src + i);
        v4h o; o[0] = (_Float16)v.x; o[1] = (_Float16)v.y;
               o[2] = (_Float16)v.z; o[3] = (_Float16)v.w;
        *(v4h*)(dst + i) = o;
    }
}

// ---------------------------------------------------------------------------
// Wt[n][k] = (f16) W[k][n] : LDS-tiled 32x32 transpose + convert.
// grid (K/32, N/32), block (32,8)
// ---------------------------------------------------------------------------
__global__ __launch_bounds__(256)
void transpose_cvt_f16_kernel(const float* __restrict__ W, _Float16* __restrict__ Wt,
                              int K, int N)
{
    __shared__ float t[32][33];
    const int k0 = blockIdx.x * 32, n0 = blockIdx.y * 32;
    const int tx = threadIdx.x, ty = threadIdx.y;     // 32 x 8
    #pragma unroll
    for (int r = 0; r < 32; r += 8)
        t[ty + r][tx] = W[(long)(k0 + ty + r) * N + (n0 + tx)];
    __syncthreads();
    #pragma unroll
    for (int r = 0; r < 32; r += 8)
        Wt[(long)(n0 + ty + r) * K + (k0 + tx)] = (_Float16)t[tx][ty + r];
}

// ---------------------------------------------------------------------------
// Batched C[z,M,Nn] = act(A[z,M,K] @ W + bias), A f16 row-major [M,K],
// weights pre-transposed f16 Wt[Nn][K]. All fragment loads are contiguous b128.
// 1 wave -> 16(M) x 64(N); block = 4 waves on 4 M tiles; grid (M/64, Nn/64, B).
// K-loop: ping-pong 2-stage pipeline (unroll 64), with an explicit
// sched_group_barrier pipeline [10 loads][4 wmma][10 loads][4 wmma] so the
// scheduler keeps a full stage of loads in flight behind every WMMA group
// instead of sinking loads to their uses (which produced s_wait_loadcnt 0x0).
// Requires K % 64 == 0 (K = 512 or 256 here).
// Writes f32 (C32) and/or f16 (C16) outputs.
// ---------------------------------------------------------------------------
__global__ __launch_bounds__(128, 1)
void wmma_gemm_f16(const _Float16* __restrict__ A, const _Float16* __restrict__ Wt,
                   const float* __restrict__ bias, float* __restrict__ C32,
                   _Float16* __restrict__ C16,
                   int M, int K, int Nn, long strideA, long strideC, int relu)
{
    const int lane = threadIdx.x & 31;
    const int wave = threadIdx.x >> 5;
    const int m0   = (blockIdx.x * 4 + wave) * 16;
    const int n0   = blockIdx.y * 64;
    const int z    = blockIdx.z;

    const int mrow = m0 + (lane & 15);
    const int kA   = (lane >> 4) << 4;   // 0 or 16
    const int ncol = lane & 15;
    const int kB   = (lane >> 4) << 4;   // 0 or 16

    const _Float16* ap = A + (long)z * strideA + (long)mrow * K + kA;
    const _Float16* bp = Wt + (long)(n0 + ncol) * K + kB;
    const long bs = (long)16 * K;        // stride between the 4 n-tiles

    v8f acc[4] = {v8f{}, v8f{}, v8f{}, v8f{}};

    // ---- stage 0 prologue (k = 0) ----
    v16h a0 = load_afrag(ap);
    v16h b0[4];
    #pragma unroll
    for (int t = 0; t < 4; ++t) b0[t] = load_bfrag(bp + t * bs);

    v16h a1, b1[4];

    // ---- steady state: two 32-wide stages per iteration, no copies ----
    int k = 0;
    for (; k < K - 64; k += 64) {
        a1 = load_afrag(ap + k + 32);
        #pragma unroll
        for (int t = 0; t < 4; ++t) b1[t] = load_bfrag(bp + t * bs + k + 32);
        #pragma unroll
        for (int t = 0; t < 4; ++t)
            acc[t] = __builtin_amdgcn_wmma_f32_16x16x32_f16(
                false, a0, false, b0[t], (short)0, acc[t], false, false);

        a0 = load_afrag(ap + k + 64);
        #pragma unroll
        for (int t = 0; t < 4; ++t) b0[t] = load_bfrag(bp + t * bs + k + 64);
        #pragma unroll
        for (int t = 0; t < 4; ++t)
            acc[t] = __builtin_amdgcn_wmma_f32_16x16x32_f16(
                false, a1, false, b1[t], (short)0, acc[t], false, false);

        // scheduling pipeline: loads stay a full stage ahead of the WMMAs
        SCHED_GROUP(0x0020, 10, 0);   // VMEM reads: a1 + b1[0..3]  (2 b128 each)
        SCHED_GROUP(0x0008,  4, 0);   // WMMA stage 0
        SCHED_GROUP(0x0020, 10, 0);   // VMEM reads: a0 + b0[0..3]
        SCHED_GROUP(0x0008,  4, 0);   // WMMA stage 1
    }

    // ---- tail pair (k == K-64) ----
    a1 = load_afrag(ap + K - 32);
    #pragma unroll
    for (int t = 0; t < 4; ++t) b1[t] = load_bfrag(bp + t * bs + K - 32);
    #pragma unroll
    for (int t = 0; t < 4; ++t)
        acc[t] = __builtin_amdgcn_wmma_f32_16x16x32_f16(
            false, a0, false, b0[t], (short)0, acc[t], false, false);
    #pragma unroll
    for (int t = 0; t < 4; ++t)
        acc[t] = __builtin_amdgcn_wmma_f32_16x16x32_f16(
            false, a1, false, b1[t], (short)0, acc[t], false, false);

    // C layout: VGPR r -> row = m0 + 8*(lane>=16) + r, col = lane&15 (+ tile offset)
    const int  mbase = m0 + ((lane >> 4) << 3);
    const long cb    = (long)z * strideC;
    #pragma unroll
    for (int t = 0; t < 4; ++t) {
        const int n = n0 + t * 16 + ncol;
        const float bz = bias ? bias[n] : 0.0f;
        #pragma unroll
        for (int r = 0; r < 8; ++r) {
            float v = acc[t][r] + bz;
            if (relu) v = fmaxf(v, 0.0f);
            const long idx = cb + (long)(mbase + r) * Nn + n;
            if (C32) C32[idx] = v;
            if (C16) C16[idx] = (_Float16)v;
        }
    }
}

// ---------------------------------------------------------------------------
// out[z,i,j] = sigmoid( sum_h relu(a[z,i,h] + c[z,j,h] + b1[h]) * w2[h] + b2 )
// 16x16 (i,j) tile per 256-thread block; rows staged in padded LDS, float4 loop.
// Optionally also writes an f16 copy (for downstream WMMA consumption).
// ---------------------------------------------------------------------------
__global__ __launch_bounds__(256)
void pair_scores_kernel(const float* __restrict__ a, const float* __restrict__ c,
                        const float* __restrict__ b1, const float* __restrict__ w2,
                        const float* __restrict__ b2s, float* __restrict__ out,
                        _Float16* __restrict__ out16)
{
    __shared__ float sA[16 * PS_PAD];
    __shared__ float sC[16 * PS_PAD];
    __shared__ float sB1[HDIM];
    __shared__ float sW2[HDIM];

    const int z   = blockIdx.z;
    const int i0  = blockIdx.x * 16;
    const int j0  = blockIdx.y * 16;
    const int tid = threadIdx.x;

    const float* ab = a + ((long)z * NNODE + i0) * HDIM;
    const float* cb = c + ((long)z * NNODE + j0) * HDIM;
    for (int idx = tid; idx < 16 * HDIM; idx += 256) {
        const int r = idx >> 8;          // HDIM == 256
        const int h = idx & 255;
        sA[r * PS_PAD + h] = ab[(long)r * HDIM + h];
        sC[r * PS_PAD + h] = cb[(long)r * HDIM + h];
    }
    sB1[tid] = b1[tid];
    sW2[tid] = w2[tid];
    __syncthreads();

    const int il = tid >> 4;
    const int jl = tid & 15;
    const float4* pa4 = (const float4*)&sA[il * PS_PAD];
    const float4* pc4 = (const float4*)&sC[jl * PS_PAD];
    const float4* pb4 = (const float4*)sB1;
    const float4* pw4 = (const float4*)sW2;

    float s = 0.0f;
    #pragma unroll 4
    for (int h4 = 0; h4 < HDIM / 4; ++h4) {
        const float4 av = pa4[h4], cv = pc4[h4], bv = pb4[h4], wv = pw4[h4];
        s = fmaf(fmaxf(av.x + cv.x + bv.x, 0.0f), wv.x, s);
        s = fmaf(fmaxf(av.y + cv.y + bv.y, 0.0f), wv.y, s);
        s = fmaf(fmaxf(av.z + cv.z + bv.z, 0.0f), wv.z, s);
        s = fmaf(fmaxf(av.w + cv.w + bv.w, 0.0f), wv.w, s);
    }
    const float r = sigmoidf_(s + b2s[0]);
    const long  o = ((long)z * NNODE + (i0 + il)) * NNODE + (j0 + jl);
    out[o] = r;
    if (out16) out16[o] = (_Float16)r;
}

// ---------------------------------------------------------------------------
// invariance = 1 - (mean|c[b]-c[b-1 mod B]| + mean_ij std_b(ddof=1)) / 2
// ---------------------------------------------------------------------------
__global__ void zero2_kernel(float* acc) { if (threadIdx.x < 2) acc[threadIdx.x] = 0.0f; }

__global__ __launch_bounds__(256)
void invariance_reduce_kernel(const float* __restrict__ causal, float* __restrict__ acc)
{
    const int p = blockIdx.x * 256 + threadIdx.x;    // (i,j) pair index in [0, N*N)
    float x[BATCH];
    float mean = 0.0f;
    #pragma unroll
    for (int b = 0; b < BATCH; ++b) {
        x[b] = causal[(long)b * (NNODE * NNODE) + p];
        mean += x[b];
    }
    mean *= (1.0f / BATCH);
    float sabs = 0.0f, svar = 0.0f;
    #pragma unroll
    for (int b = 0; b < BATCH; ++b) {
        sabs += fabsf(x[b] - x[(b + BATCH - 1) % BATCH]);   // roll(+1, axis=0)
        const float d = x[b] - mean;
        svar = fmaf(d, d, svar);
    }
    const float stdv = sqrtf(svar * (1.0f / (BATCH - 1)));  // ddof=1

    __shared__ float rA[256], rS[256];
    rA[threadIdx.x] = sabs; rS[threadIdx.x] = stdv;
    __syncthreads();
    for (int off = 128; off > 0; off >>= 1) {
        if (threadIdx.x < off) {
            rA[threadIdx.x] += rA[threadIdx.x + off];
            rS[threadIdx.x] += rS[threadIdx.x + off];
        }
        __syncthreads();
    }
    if (threadIdx.x == 0) {
        atomicAdd(&acc[0], rA[0]);
        atomicAdd(&acc[1], rS[0]);
    }
}

__global__ void finalize_kernel(const float* __restrict__ acc, float* __restrict__ out)
{
    if (threadIdx.x == 0) {
        const float stability   = acc[0] / (float)(BATCH * NNODE * NNODE);
        const float consistency = acc[1] / (float)(NNODE * NNODE);
        out[0] = 1.0f - (stability + consistency) * 0.5f;
    }
}

// ---------------------------------------------------------------------------
extern "C" void kernel_launch(void* const* d_in, const int* in_sizes, int n_in,
                              void* d_out, int out_size, void* d_ws, size_t ws_size,
                              hipStream_t stream)
{
    (void)in_sizes; (void)n_in; (void)out_size; (void)ws_size;

    const float* img   = (const float*)d_in[0];
    const float* txt   = (const float*)d_in[1];
    const float* ftW1  = (const float*)d_in[2];
    const float* ftb1  = (const float*)d_in[3];
    const float* ftW2  = (const float*)d_in[4];
    const float* ftb2  = (const float*)d_in[5];
    const float* slW1a = (const float*)d_in[6];
    const float* slW1b = (const float*)d_in[7];
    const float* slb1  = (const float*)d_in[8];
    const float* slW2  = (const float*)d_in[9];
    const float* slb2  = (const float*)d_in[10];
    const float* cnW1a = (const float*)d_in[11];
    const float* cnW1b = (const float*)d_in[12];
    const float* cnb1  = (const float*)d_in[13];
    const float* cnW2  = (const float*)d_in[14];
    const float* cnb2  = (const float*)d_in[15];

    const long nBNF = (long)BATCH * NNODE * FDIM;    // 1,048,576
    const long nBNH = (long)BATCH * NNODE * HDIM;    //   524,288

    float* structure = (float*)d_out;                // [B,N,N] f32
    float* causal    = structure + nBNH;             // [B,N,N] f32
    float* invar     = causal + nBNH;                // scalar

    // ---- workspace layout (all chunks 16B-aligned: power-of-two counts) ----
    _Float16* p16 = (_Float16*)d_ws;
    _Float16* imgA16  = p16;  p16 += nBNF;           // img features f16
    _Float16* txtA16  = p16;  p16 += nBNF;           // txt features f16
    _Float16* wtFtW1  = p16;  p16 += (long)FDIM * HDIM;
    _Float16* wtFtW2  = p16;  p16 += (long)HDIM * HDIM;
    _Float16* wtSlW1a = p16;  p16 += (long)HDIM * HDIM;
    _Float16* wtSlW1b = p16;  p16 += (long)HDIM * HDIM;
    _Float16* wtCnW1a = p16;  p16 += (long)HDIM * HDIM;
    _Float16* wtCnW1b = p16;  p16 += (long)HDIM * HDIM;
    _Float16* h1_16   = p16;  p16 += nBNH;           // MLP hidden (reused img/txt)
    _Float16* imgh16  = p16;  p16 += nBNH;
    _Float16* txth16  = p16;  p16 += nBNH;
    _Float16* struc16 = p16;  p16 += nBNH;
    float*    aF32    = (float*)p16;                 // pair-score 'a' (and a2)
    float*    cF32    = aF32 + nBNH;                 // pair-score 'c' (and c2)
    float*    acc     = cF32 + nBNH;                 // 2 floats

    const dim3 bGemm(128), gGemm(NNODE / 64, HDIM / 64, BATCH);
    const dim3 bPair(256), gPair(NNODE / 16, NNODE / 16, BATCH);
    const long sNF = (long)NNODE * FDIM, sNH = (long)NNODE * HDIM;

    // ---- precision staging: inputs + weights -> f16 (weights transposed) ----
    cvt_f16_kernel<<<dim3(nBNF / 4 / 256), 256, 0, stream>>>(img, imgA16, nBNF);
    cvt_f16_kernel<<<dim3(nBNF / 4 / 256), 256, 0, stream>>>(txt, txtA16, nBNF);
    const dim3 bT(32, 8);
    transpose_cvt_f16_kernel<<<dim3(FDIM / 32, HDIM / 32), bT, 0, stream>>>(ftW1,  wtFtW1,  FDIM, HDIM);
    transpose_cvt_f16_kernel<<<dim3(HDIM / 32, HDIM / 32), bT, 0, stream>>>(ftW2,  wtFtW2,  HDIM, HDIM);
    transpose_cvt_f16_kernel<<<dim3(HDIM / 32, HDIM / 32), bT, 0, stream>>>(slW1a, wtSlW1a, HDIM, HDIM);
    transpose_cvt_f16_kernel<<<dim3(HDIM / 32, HDIM / 32), bT, 0, stream>>>(slW1b, wtSlW1b, HDIM, HDIM);
    transpose_cvt_f16_kernel<<<dim3(HDIM / 32, HDIM / 32), bT, 0, stream>>>(cnW1a, wtCnW1a, HDIM, HDIM);
    transpose_cvt_f16_kernel<<<dim3(HDIM / 32, HDIM / 32), bT, 0, stream>>>(cnW1b, wtCnW1b, HDIM, HDIM);

    // ---- feature_transform(image / text) ----
    wmma_gemm_f16<<<gGemm, bGemm, 0, stream>>>(imgA16, wtFtW1, ftb1, nullptr, h1_16,
        NNODE, FDIM, HDIM, sNF, sNH, 1);
    wmma_gemm_f16<<<gGemm, bGemm, 0, stream>>>(h1_16, wtFtW2, ftb2, nullptr, imgh16,
        NNODE, HDIM, HDIM, sNH, sNH, 0);
    wmma_gemm_f16<<<gGemm, bGemm, 0, stream>>>(txtA16, wtFtW1, ftb1, nullptr, h1_16,
        NNODE, FDIM, HDIM, sNF, sNH, 1);
    wmma_gemm_f16<<<gGemm, bGemm, 0, stream>>>(h1_16, wtFtW2, ftb2, nullptr, txth16,
        NNODE, HDIM, HDIM, sNH, sNH, 0);

    // ---- structure_learning: a = img_h@W1a, c = txt_h@W1b; b1 folded in pairwise ----
    wmma_gemm_f16<<<gGemm, bGemm, 0, stream>>>(imgh16, wtSlW1a, nullptr, aF32, nullptr,
        NNODE, HDIM, HDIM, sNH, sNH, 0);
    wmma_gemm_f16<<<gGemm, bGemm, 0, stream>>>(txth16, wtSlW1b, nullptr, cF32, nullptr,
        NNODE, HDIM, HDIM, sNH, sNH, 0);
    pair_scores_kernel<<<gPair, bPair, 0, stream>>>(aF32, cF32, slb1, slW2, slb2,
        structure, struc16);

    // ---- causal_net on structure ([B,N,H] role, H==N) ----
    wmma_gemm_f16<<<gGemm, bGemm, 0, stream>>>(struc16, wtCnW1a, nullptr, aF32, nullptr,
        NNODE, HDIM, HDIM, sNH, sNH, 0);
    wmma_gemm_f16<<<gGemm, bGemm, 0, stream>>>(struc16, wtCnW1b, nullptr, cF32, nullptr,
        NNODE, HDIM, HDIM, sNH, sNH, 0);
    pair_scores_kernel<<<gPair, bPair, 0, stream>>>(aF32, cF32, cnb1, cnW2, cnb2,
        causal, nullptr);

    // ---- verify_causal_invariance ----
    zero2_kernel<<<1, 32, 0, stream>>>(acc);
    invariance_reduce_kernel<<<(NNODE * NNODE) / 256, 256, 0, stream>>>(causal, acc);
    finalize_kernel<<<1, 32, 0, stream>>>(acc, invar);
}
